// Surrogate_13975823581279
// MI455X (gfx1250) — matrix operand order repaired and checked by reading, not compiled, not run on hardware
//
#include <hip/hip_runtime.h>
#include <hip/hip_bf16.h>

// ---------------------------------------------------------------------------
// MI455X (gfx1250) implementation.
//   fp16 V_WMMA_F32_16X16X32_F16 GEMM core, tiles staged via the Tensor Data
//   Mover (tensor_load_to_lds) with TENSORcnt synchronization.
// Workspace layout (fp16 elements), ~252 MB total for T=10,B=2048,D=1024,H=2048:
//   zt   [T][B][D]  masked fp16 copy of z per type
//   W1t  [T][H][D]  = W1^T  (rows = n, contiguous along k)
//   Wmut [T][D][H]  = Wmu^T
//   Wsgt [T][D][H]  = Wsig^T
//   e    [T][B][H]  hidden activations
// ---------------------------------------------------------------------------

typedef __attribute__((ext_vector_type(4)))  unsigned int uint32x4;
typedef __attribute__((ext_vector_type(8)))  int          int32x8;
typedef __attribute__((ext_vector_type(4)))  int          int32x4;
typedef __attribute__((ext_vector_type(8)))  _Float16     v8h;
typedef __attribute__((ext_vector_type(16))) _Float16     v16h;
typedef __attribute__((ext_vector_type(8)))  float        v8f;

#define BM 128
#define BN 128
#define BK 32
#define LDT 40                  // padded LDS row pitch in fp16 elems (64B data + 16B pad)
#define TILE_H16 (BM * LDT)     // fp16 elems per LDS tile buffer

__device__ __forceinline__ _Float16 f2h(float f) { return (_Float16)f; }

// Low 32 bits of a generic pointer to LDS == byte offset within the
// workgroup's LDS allocation (ISA 10.2 aperture mapping).
__device__ __forceinline__ unsigned lds_offset(const void* p) {
  return (unsigned)(uintptr_t)p;
}

// ---- TDM: copy a rows x row_elems (fp16) tile, row stride row_stride_elems,
// from global to LDS, inserting 16B of pad after every 64B (-> 80B LDS pitch).
__device__ __forceinline__ void tdm_load_tile(unsigned lds_byte_off, const void* gp,
                                              unsigned rows, unsigned row_elems,
                                              unsigned long long row_stride_elems) {
  unsigned long long ga = (unsigned long long)(uintptr_t)gp;
  uint32x4 g0;
  g0[0] = 1u;                                    // count=1, user mode
  g0[1] = lds_byte_off;                          // lds_addr (bytes)
  g0[2] = (unsigned)(ga & 0xFFFFFFFFull);        // global_addr[31:0]
  g0[3] = ((unsigned)(ga >> 32) & 0x01FFFFFFu)   // global_addr[56:32]
          | (2u << 30);                          // descriptor type = 2 ("image")
  unsigned w0 = (1u << 16)    // data_size = 2 bytes
              | (1u << 20)    // pad_enable
              | (3u << 22)    // pad_interval: 16 DWORDs (64B) between pads
              | (3u << 25);   // pad_amount: 4 DWORDs (16B)
  int32x8 g1;
  g1[0] = (int)w0;
  g1[1] = (int)((row_elems & 0xFFFFu) << 16);                   // tensor_dim0[15:0]
  g1[2] = (int)((row_elems >> 16) | ((rows & 0xFFFFu) << 16));  // tensor_dim0 hi | tensor_dim1 lo
  g1[3] = (int)((rows >> 16) | (row_elems << 16));              // tensor_dim1 hi | tile_dim0
  g1[4] = (int)(rows & 0xFFFFu);                                // tile_dim1 | tile_dim2=0
  g1[5] = (int)(row_stride_elems & 0xFFFFFFFFull);              // tensor_dim0_stride[31:0]
  g1[6] = (int)((row_stride_elems >> 32) & 0xFFFFull);          // stride0[47:32] | stride1=0
  g1[7] = 0;
  int32x4 g2 = {0, 0, 0, 0};
  int32x4 g3 = {0, 0, 0, 0};
#if __clang_major__ >= 23
  int32x8 g4 = {0, 0, 0, 0, 0, 0, 0, 0};
  __builtin_amdgcn_tensor_load_to_lds(g0, g1, g2, g3, g4, 0);
#else
  __builtin_amdgcn_tensor_load_to_lds(g0, g1, g2, g3, 0);
#endif
}

// ---- fragment loads (per-lane layouts per CDNA5 ISA 7.12.2) ----
// A operand (16x32, M=lane%16): lane holds K in [hi*8,hi*8+8) U [16+hi*8, ...+8)
__device__ __forceinline__ v16h load_frag_a(const _Float16* row_base, int hi) {
  v8h lo = *(const v8h*)(row_base + hi * 8);
  v8h up = *(const v8h*)(row_base + 16 + hi * 8);
  return __builtin_shufflevector(lo, up, 0, 1, 2, 3, 4, 5, 6, 7,
                                 8, 9, 10, 11, 12, 13, 14, 15);
}
// B operand (32x16, N=lane%16): lane holds K in [hi*8,hi*8+8) U [16+hi*8, ...+8)
__device__ __forceinline__ v16h load_frag_b(const _Float16* col_base, int hi) {
  v8h lo = *(const v8h*)(col_base + hi * 8);
  v8h up = *(const v8h*)(col_base + 16 + hi * 8);
  return __builtin_shufflevector(lo, up, 0, 1, 2, 3, 4, 5, 6, 7,
                                 8, 9, 10, 11, 12, 13, 14, 15);
}

// ---------------------------------------------------------------------------
// Prep kernels
// ---------------------------------------------------------------------------
__global__ __launch_bounds__(256) void prep_zt(const float* __restrict__ z,
                                               const long long* __restrict__ tids,
                                               _Float16* __restrict__ zt,
                                               int B, int D, int T) {
  size_t i = ((size_t)blockIdx.x * 256 + threadIdx.x) * 8;
  size_t total = (size_t)T * B * D;
  if (i >= total) return;
  int d = (int)(i % D);
  size_t bd = i / D;
  int b = (int)(bd % B);
  int t = (int)(bd / B);
  const float* zp = z + (size_t)b * D + d;
  const long long* tp = tids + d;
  v8h o;
#pragma unroll
  for (int j = 0; j < 8; ++j)
    o[j] = (tp[j] == (long long)t) ? f2h(zp[j]) : (_Float16)0.0f;
  *(v8h*)(zt + i) = o;
}

// dst[t][c][r] = fp16(src[t][r][c]); R, C multiples of 32
__global__ __launch_bounds__(256) void transpose_cvt(const float* __restrict__ src,
                                                     _Float16* __restrict__ dst,
                                                     int R, int C) {
  __shared__ float tile[32][33];
  const int t = blockIdx.z;
  const int r0 = blockIdx.y * 32;
  const int c0 = blockIdx.x * 32;
  const float* s = src + (size_t)t * R * C;
  _Float16* d = dst + (size_t)t * R * C;
  const int x = threadIdx.x & 31;
  const int y0 = (threadIdx.x >> 5) * 4;
#pragma unroll
  for (int i = 0; i < 4; ++i)
    tile[y0 + i][x] = s[(size_t)(r0 + y0 + i) * C + (c0 + x)];
  __syncthreads();
#pragma unroll
  for (int i = 0; i < 4; ++i)
    d[(size_t)(c0 + y0 + i) * R + (r0 + x)] = f2h(tile[x][y0 + i]);
}

// ---------------------------------------------------------------------------
// GEMM1: e[t] = relu(zt[t] @ W1[t] + b1[t])    (M=B, N=H, K=D)
// ---------------------------------------------------------------------------
__global__ __launch_bounds__(256) void gemm1_e(const _Float16* __restrict__ zt,
                                               const _Float16* __restrict__ w1t,
                                               const float* __restrict__ b1,
                                               _Float16* __restrict__ e,
                                               int B, int D, int H) {
  __shared__ _Float16 sA[2][TILE_H16];
  __shared__ _Float16 sB[2][TILE_H16];
  const int t = blockIdx.z;
  const int m0 = blockIdx.y * BM;
  const int n0 = blockIdx.x * BN;
  const _Float16* Ag = zt + ((size_t)t * B + m0) * D;   // row stride D
  const _Float16* Bg = w1t + ((size_t)t * H + n0) * D;  // W1^T rows (n), stride D
  const int kiters = D / BK;
  const int wave = threadIdx.x >> 5;
  const int lane = threadIdx.x & 31;
  const int wm = wave >> 1, wn = wave & 1;
  const int lr = lane & 15, hi = lane >> 4;
  const bool loader = (wave == 0);

  v8f acc[2][4] = {};
  if (loader) {
    tdm_load_tile(lds_offset(&sA[0][0]), Ag, BM, BK, (unsigned long long)D);
    tdm_load_tile(lds_offset(&sB[0][0]), Bg, BN, BK, (unsigned long long)D);
  }
  for (int it = 0; it < kiters; ++it) {
    const int cur = it & 1;
    if (loader) {
      if (it + 1 < kiters) {
        tdm_load_tile(lds_offset(&sA[cur ^ 1][0]), Ag + (size_t)(it + 1) * BK, BM, BK,
                      (unsigned long long)D);
        tdm_load_tile(lds_offset(&sB[cur ^ 1][0]), Bg + (size_t)(it + 1) * BK, BN, BK,
                      (unsigned long long)D);
        __builtin_amdgcn_s_wait_tensorcnt(2);  // tiles for iteration `it` done
      } else {
        __builtin_amdgcn_s_wait_tensorcnt(0);
      }
    }
    __syncthreads();
    v16h af[2];
#pragma unroll
    for (int im = 0; im < 2; ++im) {
      const int row = wm * 32 + im * 16 + lr;
      af[im] = load_frag_a(&sA[cur][row * LDT], hi);
    }
#pragma unroll
    for (int in = 0; in < 4; ++in) {
      const int col = wn * 64 + in * 16 + lr;
      v16h bf = load_frag_b(&sB[cur][col * LDT], hi);
#pragma unroll
      for (int im = 0; im < 2; ++im)
        acc[im][in] = __builtin_amdgcn_wmma_f32_16x16x32_f16(
            false, af[im], false, bf, (short)0, acc[im][in], false, false);
    }
    __syncthreads();  // protect LDS before next iteration's TDM overwrite
  }
  // epilogue: relu(acc + b1) -> fp16 e[t][b][h]
#pragma unroll
  for (int in = 0; in < 4; ++in) {
    const int gh = n0 + wn * 64 + in * 16 + lr;
    const float bias = b1[(size_t)t * H + gh];
#pragma unroll
    for (int im = 0; im < 2; ++im) {
      const int gb0 = m0 + wm * 32 + im * 16 + hi * 8;
#pragma unroll
      for (int r = 0; r < 8; ++r) {
        float v = acc[im][in][r] + bias;
        v = v > 0.0f ? v : 0.0f;
        e[((size_t)t * B + gb0 + r) * H + gh] = f2h(v);
      }
    }
  }
}

// ---------------------------------------------------------------------------
// GEMM2: mu = e @ Wmu + bmu ; sigma = softplus(e @ Wsig + bsig) + eta
//        (M=B, N=D, K=H)
// ---------------------------------------------------------------------------
__global__ __launch_bounds__(256) void gemm2_out(const _Float16* __restrict__ e,
                                                 const _Float16* __restrict__ wmut,
                                                 const _Float16* __restrict__ wsgt,
                                                 const float* __restrict__ bmu,
                                                 const float* __restrict__ bsig,
                                                 float* __restrict__ mu_out,
                                                 float* __restrict__ sig_out,
                                                 int B, int D, int H) {
  __shared__ _Float16 sA[2][TILE_H16];
  __shared__ _Float16 sBm[2][TILE_H16];
  __shared__ _Float16 sBs[2][TILE_H16];
  const int t = blockIdx.z;
  const int m0 = blockIdx.y * BM;
  const int n0 = blockIdx.x * BN;
  const _Float16* Ag = e + ((size_t)t * B + m0) * H;      // row stride H
  const _Float16* Bmg = wmut + ((size_t)t * D + n0) * H;  // Wmu^T rows (d), stride H
  const _Float16* Bsg = wsgt + ((size_t)t * D + n0) * H;
  const int kiters = H / BK;
  const int wave = threadIdx.x >> 5;
  const int lane = threadIdx.x & 31;
  const int wm = wave >> 1, wn = wave & 1;
  const int lr = lane & 15, hi = lane >> 4;
  const bool loader = (wave == 0);

  v8f am[2][4] = {};
  v8f as[2][4] = {};
  if (loader) {
    tdm_load_tile(lds_offset(&sA[0][0]), Ag, BM, BK, (unsigned long long)H);
    tdm_load_tile(lds_offset(&sBm[0][0]), Bmg, BN, BK, (unsigned long long)H);
    tdm_load_tile(lds_offset(&sBs[0][0]), Bsg, BN, BK, (unsigned long long)H);
  }
  for (int it = 0; it < kiters; ++it) {
    const int cur = it & 1;
    if (loader) {
      if (it + 1 < kiters) {
        const size_t ko = (size_t)(it + 1) * BK;
        tdm_load_tile(lds_offset(&sA[cur ^ 1][0]), Ag + ko, BM, BK, (unsigned long long)H);
        tdm_load_tile(lds_offset(&sBm[cur ^ 1][0]), Bmg + ko, BN, BK, (unsigned long long)H);
        tdm_load_tile(lds_offset(&sBs[cur ^ 1][0]), Bsg + ko, BN, BK, (unsigned long long)H);
        __builtin_amdgcn_s_wait_tensorcnt(3);
      } else {
        __builtin_amdgcn_s_wait_tensorcnt(0);
      }
    }
    __syncthreads();
    v16h af[2];
#pragma unroll
    for (int im = 0; im < 2; ++im) {
      const int row = wm * 32 + im * 16 + lr;
      af[im] = load_frag_a(&sA[cur][row * LDT], hi);
    }
#pragma unroll
    for (int in = 0; in < 4; ++in) {
      const int col = wn * 64 + in * 16 + lr;
      v16h bfm = load_frag_b(&sBm[cur][col * LDT], hi);
      v16h bfs = load_frag_b(&sBs[cur][col * LDT], hi);
#pragma unroll
      for (int im = 0; im < 2; ++im) {
        am[im][in] = __builtin_amdgcn_wmma_f32_16x16x32_f16(
            false, af[im], false, bfm, (short)0, am[im][in], false, false);
        as[im][in] = __builtin_amdgcn_wmma_f32_16x16x32_f16(
            false, af[im], false, bfs, (short)0, as[im][in], false, false);
      }
    }
    __syncthreads();
  }
  // epilogue
#pragma unroll
  for (int in = 0; in < 4; ++in) {
    const int gd = n0 + wn * 64 + in * 16 + lr;
    const float bm = bmu[(size_t)t * D + gd];
    const float bs = bsig[(size_t)t * D + gd];
#pragma unroll
    for (int im = 0; im < 2; ++im) {
      const int gb0 = m0 + wm * 32 + im * 16 + hi * 8;
#pragma unroll
      for (int r = 0; r < 8; ++r) {
        const size_t oidx = ((size_t)t * B + gb0 + r) * D + gd;
        mu_out[oidx] = am[im][in][r] + bm;
        const float x = as[im][in][r] + bs;
        // stable softplus: max(x,0) + log1p(exp(-|x|))
        sig_out[oidx] = fmaxf(x, 0.0f) + log1pf(expf(-fabsf(x))) + 1e-6f;
      }
    }
  }
}

// ---------------------------------------------------------------------------
// Host launcher
// ---------------------------------------------------------------------------
extern "C" void kernel_launch(void* const* d_in, const int* in_sizes, int n_in,
                              void* d_out, int out_size, void* d_ws, size_t ws_size,
                              hipStream_t stream) {
  const float* z = (const float*)d_in[0];
  const long long* tids = (const long long*)d_in[1];  // int64 type ids
  const float* W1 = (const float*)d_in[2];
  const float* b1 = (const float*)d_in[3];
  const float* Wmu = (const float*)d_in[4];
  const float* bmu = (const float*)d_in[5];
  const float* Wsig = (const float*)d_in[6];
  const float* bsig = (const float*)d_in[7];
  float* out = (float*)d_out;

  const int D = in_sizes[1];          // t is [D,1]
  const int T = in_sizes[5] / D;      // bmu is [T,D]
  const int B = in_sizes[0] / D;      // z is [B,D]
  const int H = in_sizes[3] / T;      // b1 is [T,H]

  // workspace carve-up (fp16 elements)
  _Float16* ws = (_Float16*)d_ws;
  const size_t n_zt = (size_t)T * B * D;
  const size_t n_w = (size_t)T * D * H;
  _Float16* zt = ws;
  _Float16* w1t = zt + n_zt;
  _Float16* wmut = w1t + n_w;
  _Float16* wsgt = wmut + n_w;
  _Float16* e = wsgt + n_w;  // [T][B][H]

  // 1) masked fp16 z per type
  {
    size_t total = (size_t)T * B * D;
    dim3 g((unsigned)((total / 8 + 255) / 256));
    prep_zt<<<g, 256, 0, stream>>>(z, tids, zt, B, D, T);
  }
  // 2) transpose+convert weights: W1 [T][D][H] -> [T][H][D]; Wmu/Wsig [T][H][D] -> [T][D][H]
  {
    dim3 g1(H / 32, D / 32, T);
    transpose_cvt<<<g1, 256, 0, stream>>>(W1, w1t, D, H);
    dim3 g2(D / 32, H / 32, T);
    transpose_cvt<<<g2, 256, 0, stream>>>(Wmu, wmut, H, D);
    transpose_cvt<<<g2, 256, 0, stream>>>(Wsig, wsgt, H, D);
  }
  // 3) e = relu(zt @ W1 + b1)
  {
    dim3 g(H / BN, B / BM, T);
    gemm1_e<<<g, 256, 0, stream>>>(zt, w1t, b1, e, B, D, H);
  }
  // 4) mu / sigma
  {
    dim3 g(D / BN, B / BM, T);
    gemm2_out<<<g, 256, 0, stream>>>(e, wmut, wsgt, bmu, bsig, out,
                                     out + (size_t)T * B * D, B, D, H);
  }
}
